// PNANet_9491877724298
// MI455X (gfx1250) — matrix operand order repaired and checked by reading, not compiled, not run on hardware
//
#include <hip/hip_runtime.h>
#include <hip/hip_bf16.h>
#include <math.h>
#include <stdint.h>

typedef __bf16 v16bf __attribute__((ext_vector_type(16)));
typedef float  v8f   __attribute__((ext_vector_type(8)));
union bf16x16 { v16bf v; uint4 u[2]; };

static constexpr int NN = 100000;     // nodes
static constexpr int NE = 1600000;    // edges
static constexpr int NG = 256;        // graphs
static constexpr int SEQ = 85;
static constexpr unsigned FLIP_NEGINF = 0x007FFFFFu;  // flip(-inf)
static constexpr unsigned FLIP_POSINF = 0xFF800000u;  // flip(+inf)
static constexpr float BN_SCALE = 0.9999950000374997f; // 1/sqrt(1+1e-5)

__device__ __forceinline__ unsigned flipf(float f) {
  unsigned u = __float_as_uint(f);
  return u ^ ((unsigned)((int)u >> 31) | 0x80000000u);
}
__device__ __forceinline__ float unflipf(unsigned u) {
  unsigned m = (u >> 31) ? 0x80000000u : 0xFFFFFFFFu;
  return __uint_as_float(u ^ m);
}
// CDNA5 async global->LDS copy (16B), tracked on ASYNCcnt.
__device__ __forceinline__ void async_g2l_b128(const __bf16* gsrc, const __bf16* ldst) {
  unsigned loff = (unsigned)(uintptr_t)ldst;
  asm volatile("global_load_async_to_lds_b128 %0, %1, off"
               :: "v"(loff), "v"(gsrc) : "memory");
}
__device__ __forceinline__ void async_wait0() {
  asm volatile("s_wait_asynccnt 0x0" ::: "memory");
}

// ---------------- utility kernels ----------------
__global__ void k_zero_f32(float* __restrict__ p, int n) {
  int i = blockIdx.x * blockDim.x + threadIdx.x;
  if (i < n) p[i] = 0.f;
}
__global__ void k_fill_u32(unsigned* __restrict__ p, int n, unsigned v) {
  int i = blockIdx.x * blockDim.x + threadIdx.x;
  if (i < n) p[i] = v;
}
__global__ void k_degree(const int* __restrict__ dst, int e, float* __restrict__ cnt) {
  int i = blockIdx.x * blockDim.x + threadIdx.x;
  if (i < e) atomicAdd(cnt + dst[i], 1.f);
}
__global__ void k_logsum(const float* __restrict__ cnt, int n, float* __restrict__ out) {
  __shared__ float sm[256];
  int i = blockIdx.x * blockDim.x + threadIdx.x;
  float v = (i < n) ? logf(cnt[i] + 1.f) : 0.f;
  sm[threadIdx.x] = v;
  __syncthreads();
  for (int s = 128; s > 0; s >>= 1) {
    if (threadIdx.x < s) sm[threadIdx.x] += sm[threadIdx.x + s];
    __syncthreads();
  }
  if (threadIdx.x == 0) atomicAdd(out, sm[0]);
}
__global__ void k_scalers(const float* __restrict__ cnt, const float* __restrict__ tot,
                          int n, float* __restrict__ amp, float* __restrict__ att) {
  int i = blockIdx.x * blockDim.x + threadIdx.x;
  if (i >= n) return;
  float avg = tot[0] / (float)n;
  float d = fmaxf(cnt[i], 1.f);
  float logd = logf(d + 1.f);
  amp[i] = logd / avg;
  att[i] = avg / logd;
}
__global__ void k_castpad(const float* __restrict__ src, int lds, int F,
                          __bf16* __restrict__ dst, int ldd, int M) {
  int i = blockIdx.x * blockDim.x + threadIdx.x;
  if (i >= M * ldd) return;
  int m = i / ldd, c = i - m * ldd;
  float v = (c < F) ? src[(size_t)m * lds + c] : 0.f;
  dst[(size_t)m * ldd + c] = (__bf16)v;
}
// fp32 W[K,Nout] -> bf16 Wt[Npad,Kpad]  (transposed, zero-padded)
__global__ void k_wprep(const float* __restrict__ w, int K, int Nout,
                        __bf16* __restrict__ wt, int Kpad, int Npad) {
  int i = blockIdx.x * blockDim.x + threadIdx.x;
  if (i >= Npad * Kpad) return;
  int n = i / Kpad, k = i - n * Kpad;
  float v = (k < K && n < Nout) ? w[(size_t)k * Nout + n] : 0.f;
  wt[(size_t)n * Kpad + k] = (__bf16)v;
}

// ---------------- edge pre_nn + aggregation (WMMA, persistent waves) ----------------
// Wi/Wj (pre-transposed bf16) staged in LDS via async copy; each wave grid-strides
// over 16-edge groups: gathers 16 node-row pairs as bf16 A fragments, accumulates
// A_dst@Wi + A_src@Wj + b with v_wmma_f32_16x16x32_bf16, then scatters to the
// segment sum/sum^2 (atomicAdd) and max/min (monotone-uint atomics).
template <int NT, int KB>
__global__ void k_edge(const __bf16* __restrict__ hb,
                       const int* __restrict__ srcI, const int* __restrict__ dstI, int E,
                       const __bf16* __restrict__ wit, const __bf16* __restrict__ wjt,
                       const float* __restrict__ bpre, int F,
                       float* __restrict__ s, float* __restrict__ s2,
                       unsigned* __restrict__ mx, unsigned* __restrict__ mn) {
  constexpr int Fpad = KB * 32;
  constexpr int NW = NT * 16 * Fpad;              // weight elements per matrix
  __shared__ __align__(16) __bf16 sWi[NW];
  __shared__ __align__(16) __bf16 sWj[NW];
  for (int i = threadIdx.x; i < NW / 8; i += blockDim.x) {
    async_g2l_b128(wit + (size_t)i * 8, sWi + (size_t)i * 8);
    async_g2l_b128(wjt + (size_t)i * 8, sWj + (size_t)i * 8);
  }
  async_wait0();
  __syncthreads();

  const int lane = threadIdx.x & 31;
  const int wv = threadIdx.x >> 5;
  const int r = lane & 15, hi = lane >> 4, kh = hi * 8;
  float bv[NT];
#pragma unroll
  for (int t = 0; t < NT; t++) {
    int c = t * 16 + r;
    bv[t] = (c < F) ? bpre[c] : 0.f;
  }

  const int wavesTotal = gridDim.x * (blockDim.x >> 5);
  const int nGroups = (E + 15) >> 4;
  for (int g = blockIdx.x * (blockDim.x >> 5) + wv; g < nGroups; g += wavesTotal) {
    const int g0 = g * 16;                        // wave-uniform
    int eA = g0 + r;
    if (eA >= E) eA = E - 1;
    const int nd = dstI[eA];
    const int ns = srcI[eA];
    const __bf16* rowD = hb + (size_t)nd * Fpad;
    const __bf16* rowS = hb + (size_t)ns * Fpad;

    v8f acc[NT];
#pragma unroll
    for (int t = 0; t < NT; t++)
#pragma unroll
      for (int j = 0; j < 8; j++) acc[t][j] = bv[t];

#pragma unroll
    for (int kb = 0; kb < KB; kb++) {
      const int kk = kb * 32;
      bf16x16 aD, aS;
      aD.u[0] = *(const uint4*)(rowD + kk + kh);
      aD.u[1] = *(const uint4*)(rowD + kk + 16 + kh);
      aS.u[0] = *(const uint4*)(rowS + kk + kh);
      aS.u[1] = *(const uint4*)(rowS + kk + 16 + kh);
#pragma unroll
      for (int t = 0; t < NT; t++) {
        const int gc = t * 16 + r;
        const __bf16* bi = sWi + (size_t)gc * Fpad + kk + kh;
        const __bf16* bj = sWj + (size_t)gc * Fpad + kk + kh;
        bf16x16 B1, B2;
        B1.u[0] = ((const uint4*)bi)[0]; B1.u[1] = ((const uint4*)bi)[2];
        B2.u[0] = ((const uint4*)bj)[0]; B2.u[1] = ((const uint4*)bj)[2];
        acc[t] = __builtin_amdgcn_wmma_f32_16x16x32_bf16(false, aD.v, false, B1.v,
                                                         (short)0, acc[t], false, false);
        acc[t] = __builtin_amdgcn_wmma_f32_16x16x32_bf16(false, aS.v, false, B2.v,
                                                         (short)0, acc[t], false, false);
      }
    }
    // D tile: element j of lane = (edge g0+j+8*hi, col t*16+(lane&15))
#pragma unroll
    for (int j = 0; j < 8; j++) {
      const int e = g0 + j + 8 * hi;
      const int node = __shfl(nd, j + 8 * hi, 32);
      const bool ok = (e < E);
#pragma unroll
      for (int t = 0; t < NT; t++) {
        const int c = t * 16 + r;
        if (ok && c < F) {
          float v = acc[t][j];
          size_t off = (size_t)node * F + c;
          atomicAdd(s + off, v);
          atomicAdd(s2 + off, v * v);
          atomicMax(mx + off, flipf(v));
          atomicMin(mn + off, flipf(v));
        }
      }
    }
  }
}

// ---------------- build Z = [h_in | agg | agg*amp | agg*att] in bf16 ----------------
__global__ void k_node_z(const float* __restrict__ hin, int ldh, int F, int Kz, int Kzpad,
                         const float* __restrict__ cnt, const float* __restrict__ amp,
                         const float* __restrict__ att,
                         const float* __restrict__ s, const float* __restrict__ s2,
                         const unsigned* __restrict__ mx, const unsigned* __restrict__ mn,
                         __bf16* __restrict__ Z, int n) {
  long idx = (long)blockIdx.x * blockDim.x + threadIdx.x;
  if (idx >= (long)n * Kzpad) return;
  int node = (int)(idx / Kzpad);
  int k = (int)(idx - (long)node * Kzpad);
  float val = 0.f;
  if (k < Kz) {
    int seg = k / F, c = k - seg * F;
    if (seg == 0) {
      val = hin[(size_t)node * ldh + c];
    } else {
      int a = (seg - 1) & 3;         // 0 mean, 1 max, 2 min, 3 std
      int grp = (seg - 1) >> 2;      // 0 plain, 1 *amp, 2 *att
      float cn = cnt[node];
      float dd = fmaxf(cn, 1.f);
      size_t off = (size_t)node * F + c;
      float v;
      if (a == 0)      v = s[off] / dd;
      else if (a == 1) v = (cn > 0.f) ? unflipf(mx[off]) : 0.f;
      else if (a == 2) v = (cn > 0.f) ? unflipf(mn[off]) : 0.f;
      else {
        float m = s[off] / dd;
        float var = s2[off] / dd - m * m;
        v = sqrtf(fmaxf(var, 0.f) + 1e-5f);
      }
      if (grp == 1) v *= amp[node];
      else if (grp == 2) v *= att[node];
      val = v;
    }
  }
  Z[(size_t)node * Kzpad + k] = (__bf16)val;
}

// ---------------- M x kpad @ kpad x 64 bf16 WMMA GEMM (Wt staged in LDS) ----------------
// post: 0 = none, 1 = BatchNorm(eval)+ReLU
__global__ void k_gemm64(const __bf16* __restrict__ A, int lda,
                         const __bf16* __restrict__ Wt, int kpad,
                         const float* __restrict__ bias, int M,
                         float* __restrict__ outF, int ldoF,
                         __bf16* __restrict__ outB, int ldoB,
                         int post, const float* __restrict__ bng,
                         const float* __restrict__ bnb) {
  extern __shared__ __align__(16) __bf16 sW[];   // [64, kpad]
  for (int i = threadIdx.x; i < (64 * kpad) / 8; i += blockDim.x)
    async_g2l_b128(Wt + (size_t)i * 8, sW + (size_t)i * 8);
  async_wait0();
  __syncthreads();

  const int lane = threadIdx.x & 31;
  const int wv = threadIdx.x >> 5;
  const int m0 = (blockIdx.x * (blockDim.x >> 5) + wv) * 16;   // wave-uniform
  if (m0 >= M) return;
  const int r = lane & 15, hi = lane >> 4, kh = hi * 8;
  int arow = m0 + r;
  if (arow >= M) arow = M - 1;
  const __bf16* Ar = A + (size_t)arow * lda;
  v8f acc[4];
#pragma unroll
  for (int t = 0; t < 4; t++) {
    float bvv = bias[t * 16 + r];
#pragma unroll
    for (int j = 0; j < 8; j++) acc[t][j] = bvv;
  }
  for (int kk = 0; kk < kpad; kk += 32) {
    bf16x16 a;
    a.u[0] = *(const uint4*)(Ar + kk + kh);
    a.u[1] = *(const uint4*)(Ar + kk + 16 + kh);
#pragma unroll
    for (int t = 0; t < 4; t++) {
      const __bf16* bp = sW + (size_t)(t * 16 + r) * kpad + kk + kh;
      bf16x16 b;
      b.u[0] = ((const uint4*)bp)[0]; b.u[1] = ((const uint4*)bp)[2];
      acc[t] = __builtin_amdgcn_wmma_f32_16x16x32_bf16(false, a.v, false, b.v,
                                                       (short)0, acc[t], false, false);
    }
  }
#pragma unroll
  for (int t = 0; t < 4; t++) {
    const int gc = t * 16 + r;
    float sc = 1.f, sh = 0.f;
    if (post == 1) { sc = bng[gc] * BN_SCALE; sh = bnb[gc]; }
#pragma unroll
    for (int j = 0; j < 8; j++) {
      const int row = m0 + j + 8 * hi;
      if (row < M) {
        float v = acc[t][j];
        if (post == 1) v = fmaxf(fmaf(v, sc, sh), 0.f);
        if (outF) outF[(size_t)row * ldoF + gc] = v;
        if (outB) outB[(size_t)row * ldoB + gc] = (__bf16)v;
      }
    }
  }
}

// ---------------- pooling + head ----------------
__global__ void k_pool(const float* __restrict__ h, const int* __restrict__ batch,
                       float* __restrict__ gsum, int n) {
  int i = blockIdx.x * blockDim.x + threadIdx.x;
  if (i >= n * 64) return;
  int node = i >> 6, c = i & 63;
  atomicAdd(gsum + (size_t)batch[node] * 64 + c, h[(size_t)node * 64 + c]);
}
__global__ void k_gcnt(const int* __restrict__ batch, float* __restrict__ gcnt, int n) {
  int i = blockIdx.x * blockDim.x + threadIdx.x;
  if (i < n) atomicAdd(gcnt + batch[i], 1.f);
}
__global__ void k_xg(const float* __restrict__ gsum, const float* __restrict__ gcnt,
                     float* __restrict__ xg) {
  int i = blockIdx.x * blockDim.x + threadIdx.x;
  if (i >= NG * 64) return;
  xg[i] = gsum[i] / fmaxf(gcnt[i >> 6], 1.f);
}
__global__ void k_fc(const float* __restrict__ A, const float* __restrict__ W,
                     const float* __restrict__ b, float* __restrict__ out,
                     int M, int K, int Nout, int relu) {
  int i = blockIdx.x * blockDim.x + threadIdx.x;
  if (i >= M * Nout) return;
  int m = i / Nout, nIdx = i - m * Nout;
  float acc = b[nIdx];
  const float* ar = A + (size_t)m * K;
  for (int k = 0; k < K; k++) acc = fmaf(ar[k], W[(size_t)k * Nout + nIdx], acc);
  if (relu) acc = fmaxf(acc, 0.f);
  out[(size_t)m * Nout + nIdx] = acc;
}
// conv1d: out[b][f][t] = sum_{c,k} emb[target[b][t+k]][c] * W[f][c][k] + bias[f]
__global__ void k_conv(const int* __restrict__ target, const float* __restrict__ emb,
                       const float* __restrict__ cw, const float* __restrict__ cb,
                       float* __restrict__ out) {
  int idx = blockIdx.x * blockDim.x + threadIdx.x;
  const int total = NG * 32 * 78;
  if (idx >= total) return;
  int t = idx % 78;
  int rem = idx / 78;
  int f = rem % 32, bg = rem / 32;
  float acc = cb[f];
  const float* wf = cw + (size_t)f * 128 * 8;
  for (int k = 0; k < 8; k++) {
    int tok = target[bg * SEQ + t + k];
    const float* er = emb + (size_t)tok * 128;
    for (int c = 0; c < 128; c++) acc = fmaf(er[c], wf[c * 8 + k], acc);
  }
  out[(size_t)bg * 2496 + f * 78 + t] = acc;
}
__global__ void k_concat(const float* __restrict__ a, const float* __restrict__ b2,
                         float* __restrict__ out, int M, int Na, int Nb) {
  int i = blockIdx.x * blockDim.x + threadIdx.x;
  int N = Na + Nb;
  if (i >= M * N) return;
  int m = i / N, c = i - m * N;
  out[i] = (c < Na) ? a[(size_t)m * Na + c] : b2[(size_t)m * Nb + (c - Na)];
}

// ---------------- host ----------------
static inline int cdiv(long a, long b) { return (int)((a + b - 1) / b); }

extern "C" void kernel_launch(void* const* d_in, const int* in_sizes, int n_in,
                              void* d_out, int out_size, void* d_ws, size_t ws_size,
                              hipStream_t stream) {
  (void)in_sizes; (void)n_in; (void)out_size; (void)ws_size;
  auto F32 = [&](int i) { return (const float*)d_in[i]; };
  const float* x = F32(0);
  struct Pna { const float *Wi, *Wj, *bpre, *Wpost, *bpost, *Wlin, *blin, *bng, *bnb; };
  auto getP = [&](int b) {
    return Pna{F32(b), F32(b + 1), F32(b + 2), F32(b + 3), F32(b + 4),
               F32(b + 5), F32(b + 6), F32(b + 7), F32(b + 8)};
  };
  Pna L[3] = {getP(1), getP(10), getP(19)};
  const float* fc1_xd_W = F32(28); const float* fc1_xd_b = F32(29);
  const float* emb = F32(30);
  const float* convW = F32(31); const float* convb = F32(32);
  const float* fc1_xt_W = F32(33); const float* fc1_xt_b = F32(34);
  const float* fc1_W = F32(35); const float* fc1_b = F32(36);
  const float* fc2_W = F32(37); const float* fc2_b = F32(38);
  const float* out_W = F32(39); const float* out_b = F32(40);
  const int* eidx = (const int*)d_in[41];
  const int* srcI = eidx;
  const int* dstI = eidx + NE;
  const int* batch = (const int*)d_in[42];
  const int* target = (const int*)d_in[43];

  char* base = (char*)d_ws;
  size_t off = 0;
  auto alloc = [&](size_t bytes) -> void* {
    off = (off + 255) & ~(size_t)255;
    void* p = base + off;
    off += bytes;
    return p;
  };
  float*    cnt   = (float*)alloc((size_t)NN * 4);
  float*    scal  = (float*)alloc(256);
  float*    amp   = (float*)alloc((size_t)NN * 4);
  float*    att   = (float*)alloc((size_t)NN * 4);
  __bf16*   xbf   = (__bf16*)alloc((size_t)NN * 96 * 2);
  __bf16*   hbf   = (__bf16*)alloc((size_t)NN * 64 * 2);
  __bf16*   t1bf  = (__bf16*)alloc((size_t)NN * 64 * 2);
  float*    h     = (float*)alloc((size_t)NN * 64 * 4);
  float*    s     = (float*)alloc((size_t)NN * 78 * 4);
  float*    s2    = (float*)alloc((size_t)NN * 78 * 4);
  unsigned* mx    = (unsigned*)alloc((size_t)NN * 78 * 4);
  unsigned* mn    = (unsigned*)alloc((size_t)NN * 78 * 4);
  __bf16*   Z     = (__bf16*)alloc((size_t)NN * 1024 * 2);
  __bf16*   wit   = (__bf16*)alloc(80 * 96 * 2);
  __bf16*   wjt   = (__bf16*)alloc(80 * 96 * 2);
  __bf16*   wpt   = (__bf16*)alloc(64 * 1024 * 2);
  __bf16*   wlt   = (__bf16*)alloc(64 * 64 * 2);
  float*    gsum  = (float*)alloc(NG * 64 * 4);
  float*    gcnt  = (float*)alloc(NG * 4);
  float*    xg    = (float*)alloc(NG * 64 * 4);
  float*    xg2   = (float*)alloc(NG * 128 * 4);
  float*    cvo   = (float*)alloc((size_t)NG * 2496 * 4);
  float*    xt    = (float*)alloc(NG * 128 * 4);
  float*    xc    = (float*)alloc(NG * 256 * 4);
  float*    a1    = (float*)alloc((size_t)NG * 1024 * 4);
  float*    a2    = (float*)alloc(NG * 256 * 4);

  const int TB = 256;
  const int EDGE_BLOCKS = 2048;   // persistent waves, grid-stride over 100K groups
  // degree + PNA scalers
  k_zero_f32<<<cdiv(NN, TB), TB, 0, stream>>>(cnt, NN);
  k_zero_f32<<<1, TB, 0, stream>>>(scal, 1);
  k_degree<<<cdiv(NE, TB), TB, 0, stream>>>(dstI, NE, cnt);
  k_logsum<<<cdiv(NN, TB), TB, 0, stream>>>(cnt, NN, scal);
  k_scalers<<<cdiv(NN, TB), TB, 0, stream>>>(cnt, scal, NN, amp, att);
  // bf16 node features for layer 1 (pad 78 -> 96)
  k_castpad<<<cdiv((long)NN * 96, TB), TB, 0, stream>>>(x, 78, 78, xbf, 96, NN);

  auto layerTail = [&](const Pna& p, const float* hin, int ldh, int F, int Kz, int Kzpad) {
    k_node_z<<<cdiv((long)NN * Kzpad, TB), TB, 0, stream>>>(
        hin, ldh, F, Kz, Kzpad, cnt, amp, att, s, s2, mx, mn, Z, NN);
    k_wprep<<<cdiv((long)64 * Kzpad, TB), TB, 0, stream>>>(p.Wpost, Kz, 64, wpt, Kzpad, 64);
    k_wprep<<<cdiv(64 * 64, TB), TB, 0, stream>>>(p.Wlin, 64, 64, wlt, 64, 64);
    // Z @ Wpost + bpost -> t1 (bf16 only, feeds next GEMM)
    k_gemm64<<<cdiv(NN, 128), TB, (size_t)64 * Kzpad * 2, stream>>>(
        Z, Kzpad, wpt, Kzpad, p.bpost, NN, nullptr, 0, t1bf, 64, 0, nullptr, nullptr);
    // t1 @ Wlin + blin -> BN+ReLU -> h (fp32) and hbf (bf16 for next edge gather)
    k_gemm64<<<cdiv(NN, 128), TB, (size_t)64 * 64 * 2, stream>>>(
        t1bf, 64, wlt, 64, p.blin, NN, h, 64, hbf, 64, 1, p.bng, p.bnb);
  };
  auto initAggr = [&](int F) {
    k_zero_f32<<<cdiv((long)NN * F, TB), TB, 0, stream>>>(s, NN * F);
    k_zero_f32<<<cdiv((long)NN * F, TB), TB, 0, stream>>>(s2, NN * F);
    k_fill_u32<<<cdiv((long)NN * F, TB), TB, 0, stream>>>(mx, NN * F, FLIP_NEGINF);
    k_fill_u32<<<cdiv((long)NN * F, TB), TB, 0, stream>>>(mn, NN * F, FLIP_POSINF);
  };

  // ---- layer 1: F=78 (Kpad 96, NT 5), Kz = 13*78 = 1014 -> 1024 ----
  k_wprep<<<cdiv(80 * 96, TB), TB, 0, stream>>>(L[0].Wi, 78, 78, wit, 96, 80);
  k_wprep<<<cdiv(80 * 96, TB), TB, 0, stream>>>(L[0].Wj, 78, 78, wjt, 96, 80);
  initAggr(78);
  k_edge<5, 3><<<EDGE_BLOCKS, TB, 0, stream>>>(xbf, srcI, dstI, NE, wit, wjt,
                                               L[0].bpre, 78, s, s2, mx, mn);
  layerTail(L[0], x, 78, 78, 1014, 1024);

  // ---- layers 2 & 3: F=64 (Kpad 64, NT 4), Kz = 832 ----
  for (int li = 1; li < 3; li++) {
    k_wprep<<<cdiv(64 * 64, TB), TB, 0, stream>>>(L[li].Wi, 64, 64, wit, 64, 64);
    k_wprep<<<cdiv(64 * 64, TB), TB, 0, stream>>>(L[li].Wj, 64, 64, wjt, 64, 64);
    initAggr(64);
    k_edge<4, 2><<<EDGE_BLOCKS, TB, 0, stream>>>(hbf, srcI, dstI, NE, wit, wjt,
                                                 L[li].bpre, 64, s, s2, mx, mn);
    layerTail(L[li], h, 64, 64, 832, 832);
  }

  // ---- global mean pool + graph head ----
  k_zero_f32<<<cdiv(NG * 64, TB), TB, 0, stream>>>(gsum, NG * 64);
  k_zero_f32<<<cdiv(NG, TB), TB, 0, stream>>>(gcnt, NG);
  k_pool<<<cdiv((long)NN * 64, TB), TB, 0, stream>>>(h, batch, gsum, NN);
  k_gcnt<<<cdiv(NN, TB), TB, 0, stream>>>(batch, gcnt, NN);
  k_xg<<<cdiv(NG * 64, TB), TB, 0, stream>>>(gsum, gcnt, xg);
  k_fc<<<cdiv(NG * 128, TB), TB, 0, stream>>>(xg, fc1_xd_W, fc1_xd_b, xg2, NG, 64, 128, 1);

  // ---- sequence branch ----
  k_conv<<<cdiv((long)NG * 32 * 78, TB), TB, 0, stream>>>(target, emb, convW, convb, cvo);
  k_fc<<<cdiv(NG * 128, TB), TB, 0, stream>>>(cvo, fc1_xt_W, fc1_xt_b, xt, NG, 2496, 128, 0);

  // ---- joint MLP ----
  k_concat<<<cdiv(NG * 256, TB), TB, 0, stream>>>(xg2, xt, xc, NG, 128, 128);
  k_fc<<<cdiv(NG * 1024, TB), TB, 0, stream>>>(xc, fc1_W, fc1_b, a1, NG, 256, 1024, 1);
  k_fc<<<cdiv(NG * 256, TB), TB, 0, stream>>>(a1, fc2_W, fc2_b, a2, NG, 1024, 256, 1);
  k_fc<<<cdiv(NG, TB), TB, 0, stream>>>(a2, out_W, out_b, (float*)d_out, NG, 256, 1, 0);
}
